// MeshDecoder_83356725280812
// MI455X (gfx1250) — matrix operand order repaired and checked by reading, not compiled, not run on hardware
//
#include <hip/hip_runtime.h>
#include <hip/hip_bf16.h>

// ---------------------------------------------------------------------------
// MeshDecoder GNN on gfx1250 (MI455X).
// Edge-MLP GEMM (213 of 216 GFLOP) runs on v_wmma_f32_16x16x32_bf16.
// Packed layer weights (73.7 KB) are staged once per workgroup into LDS
// (conflict-aware layout) and re-read per tile by persistent waves; an
// asm memory barrier per tile iteration stops LICM from hoisting the 72
// loop-invariant B-fragment ds_loads into registers (which spilled to
// scratch in the previous revision). Node features (12.8 MB bf16) stay
// L2-resident for the per-edge gathers.
// ---------------------------------------------------------------------------

typedef __attribute__((ext_vector_type(16))) __bf16 bf16x16;
typedef __attribute__((ext_vector_type(8)))  float  v8f;

#define N_NODES  50000
#define E_EDGES  800000
#define IN_DIM   256
#define HID      128
#define OUT_DIM  3
#define EDGE_DIM 4
#define N_LAYERS 4
#define MSG_IN   260          // 2*HID + EDGE_DIM
#define KBLK     9            // ceil(260/32) K-blocks of 32
#define NTILES   8            // 128 / 16 output tiles
#define WP_U4    (KBLK * NTILES * 64)   // uint4 slots per layer (73728 B)
#define EBLOCKS  1024         // persistent edge-kernel blocks (8 waves each)

// Prevents loop-invariant code motion of LDS/global loads across iterations.
#define ANTI_LICM() asm volatile("" ::: "memory")

static __device__ __forceinline__ unsigned short f2bf(float f) {
    union { float f; unsigned int u; } v; v.f = f;
    unsigned int r = v.u + 0x7FFFu + ((v.u >> 16) & 1u);   // round-to-nearest-even
    return (unsigned short)(r >> 16);
}

static __device__ __forceinline__ bf16x16 frag_from(uint4 q0, uint4 q1) {
    union { uint4 q[2]; bf16x16 v; } u;
    u.q[0] = q0; u.q[1] = q1;
    return u.v;
}

// ---------------------------------------------------------------------------
// Pack W_msg[l] (MSG_IN x HID, f32, row-major) into per-lane bf16 B-fragments:
// wp[l][kb][t][lane][j] with  col = t*16 + (lane&15),  k = kb*32 + (lane>>4)*16 + j.
// Contiguous k order (not the ISA table order), same as the A fragments (lane
// halves cover K 0..15 / 16..31). Rows k >= 260 are zero padding.
// ---------------------------------------------------------------------------
__global__ void pack_wmsg_kernel(const float* __restrict__ Wmsg,
                                 unsigned short* __restrict__ wp) {
    int gid = blockIdx.x * blockDim.x + threadIdx.x;
    if (gid >= N_LAYERS * KBLK * NTILES * 32) return;
    int lane = gid & 31;
    int t    = (gid >> 5) & 7;
    int rest = gid >> 8;            // l*KBLK + kb
    int kb   = rest % KBLK;
    int l    = rest / KBLK;
    int col  = (t << 4) + (lane & 15);
    int half = lane >> 4;
    unsigned short* out = wp + (size_t)gid * 16;
#pragma unroll
    for (int j = 0; j < 16; ++j) {
        int k = kb * 32 + half * 16 + j;
        float v = (k < MSG_IN) ? Wmsg[((size_t)l * MSG_IN + k) * HID + col] : 0.0f;
        out[j] = f2bf(v);
    }
}

// ---------------------------------------------------------------------------
// x = h @ W_in + b_in   (3.3 GF, 1.5% of total -> plain fp32 VALU is fine)
// ---------------------------------------------------------------------------
__global__ void input_proj_kernel(const float* __restrict__ h,
                                  const float* __restrict__ Win,
                                  const float* __restrict__ bin,
                                  float* __restrict__ x,
                                  unsigned short* __restrict__ xb) {
    __shared__ float hrow[IN_DIM];
    int n = blockIdx.x;
    int j = threadIdx.x;                       // 0..127
    hrow[j]       = h[(size_t)n * IN_DIM + j];
    hrow[j + 128] = h[(size_t)n * IN_DIM + j + 128];
    __syncthreads();
    float acc = bin[j];
#pragma unroll 8
    for (int k = 0; k < IN_DIM; ++k)
        acc = fmaf(hrow[k], Win[(size_t)k * HID + j], acc);
    x[(size_t)n * HID + j]  = acc;
    xb[(size_t)n * HID + j] = f2bf(acc);
}

// ---------------------------------------------------------------------------
// Edge message kernel: persistent waves, one 16-edge tile per wave iteration.
//   m = relu(concat(xb[src], xb[dst], edge_attr) @ W_msg[l] + b_msg[l])
//   agg[dst] += m   (global_atomic_add_f32)
//
// LDS slot layout for B fragments (16 B slots): slot = frag*64 + chunk*32 + lane
//   -> ds_load_b128 across 32 lanes hits each 4-bank group exactly twice
//      (optimal: a 512 B wave read needs 2 phases regardless).
// A-fragment (16x32 bf16): lane = (half,row); chunks K = koff+half*16+0..7 and
// koff+half*16+8+0..7 -> contiguous b128 loads from the gathered node row,
// all issued up front in one clause per tile.
// ---------------------------------------------------------------------------
__global__ void __launch_bounds__(256, 1)
edge_msg_kernel(const unsigned short* __restrict__ xb,
                const int*            __restrict__ eidx,     // [2][E]
                const float*          __restrict__ eattr,    // [E][4]
                const unsigned short* __restrict__ wp_layer, // [KBLK][NTILES][32][16]
                const float*          __restrict__ bmsg,     // [HID]
                float*                __restrict__ agg) {    // [N][HID]
    __shared__ uint4 ldsw[WP_U4];                            // 73728 B, < 320 KB/WGP

    // --- stage packed weights into LDS (conflict-aware slot layout) ---------
    {
        const uint4* wq = (const uint4*)wp_layer;
        for (int s = threadIdx.x; s < WP_U4; s += 256) {
            int f = s >> 6;          // fragment (kb*8 + t)
            int r = s & 63;
            int c = r >> 5;          // chunk 0/1
            int L = r & 31;          // lane
            ldsw[s] = wq[f * 64 + L * 2 + c];   // global layout: [f][lane][chunk]
        }
    }
    __syncthreads();

    const int lane = threadIdx.x & 31;
    const int wave = threadIdx.x >> 5;
    const int half = lane >> 4;
    const int row  = lane & 15;
    const int col  = lane & 15;
    const long ntiles = (E_EDGES + 15) / 16;

    // tile-invariant bias per lane (D layout: lane holds col, rows half*8+r)
    float bias[NTILES];
#pragma unroll
    for (int t = 0; t < NTILES; ++t) bias[t] = bmsg[t * 16 + col];

    for (long tile = (long)blockIdx.x * 8 + wave; tile < ntiles;
         tile += (long)EBLOCKS * 8) {
        // Keep per-tile memory ops (esp. the loop-invariant B-fragment LDS
        // reads) inside the loop; without this LICM hoists 72 fragments and
        // spills them to scratch.
        ANTI_LICM();

        const long tbase = tile * 16;
        long e = tbase + row;
        if (e >= E_EDGES) e = E_EDGES - 1;     // clamp; OOB rows never scattered
        const int s = eidx[e];
        const int d = eidx[E_EDGES + e];

        // --- front-load all A chunks for this tile (16 x b128, one clause) --
        uint4 aq[16];
        {
            const uint4* sp = (const uint4*)(xb + (size_t)s * HID);
            const uint4* dp = (const uint4*)(xb + (size_t)d * HID);
#pragma unroll
            for (int kb = 0; kb < 4; ++kb) {
                const int k4 = kb * 4;                 // uint4 offset of koff
                aq[2 * kb]           = sp[k4 + 2 * half];      // K koff+h16 .. +7
                aq[2 * kb + 1]       = sp[k4 + 2 * half + 1];  // K koff+h16+8 .. +7
                aq[2 * (kb + 4)]     = dp[k4 + 2 * half];
                aq[2 * (kb + 4) + 1] = dp[k4 + 2 * half + 1];
            }
        }
        // tail K-block: K 256..259 = edge_attr, rest zero padding
        union { bf16x16 v; uint4 q[2]; unsigned short s16[16]; } atail;
        atail.q[0] = make_uint4(0, 0, 0, 0);
        atail.q[1] = make_uint4(0, 0, 0, 0);
        if (half == 0) {
            const float4 ea = *(const float4*)(eattr + e * 4);
            atail.s16[0] = f2bf(ea.x); atail.s16[1] = f2bf(ea.y);
            atail.s16[2] = f2bf(ea.z); atail.s16[3] = f2bf(ea.w);
        }

        v8f acc[NTILES];
#pragma unroll
        for (int t = 0; t < NTILES; ++t)
#pragma unroll
            for (int r = 0; r < 8; ++r) acc[t][r] = 0.0f;

        // --- 9 K-blocks x 8 N-tiles of v_wmma_f32_16x16x32_bf16 -------------
#pragma unroll
        for (int kb = 0; kb < KBLK; ++kb) {
            const bf16x16 a = (kb < 8) ? frag_from(aq[2 * kb], aq[2 * kb + 1])
                                       : atail.v;
            const uint4* bkb = ldsw + kb * NTILES * 64;
#pragma unroll
            for (int t = 0; t < NTILES; ++t) {
                const uint4* bp = bkb + t * 64;
                const bf16x16 b = frag_from(bp[lane], bp[32 + lane]);
                acc[t] = __builtin_amdgcn_wmma_f32_16x16x32_bf16(
                    false, a, false, b, (short)0, acc[t], false, false);
            }
        }

        // --- bias + ReLU + atomic scatter to agg[dst] -----------------------
        int drow[8];
#pragma unroll
        for (int r = 0; r < 8; ++r) {
            long er = tbase + half * 8 + r;
            drow[r] = (er < E_EDGES) ? eidx[E_EDGES + er] : -1;
        }
#pragma unroll
        for (int t = 0; t < NTILES; ++t) {
            const int feat = t * 16 + col;
#pragma unroll
            for (int r = 0; r < 8; ++r) {
                if (drow[r] >= 0) {
                    float v = acc[t][r] + bias[t];
                    v = v > 0.0f ? v : 0.0f;
                    atomicAdd(agg + (size_t)drow[r] * HID + feat, v);
                }
            }
        }
    }
}

// ---------------------------------------------------------------------------
// Residual update: x += agg; refresh bf16 copy.
// ---------------------------------------------------------------------------
__global__ void update_kernel(float* __restrict__ x,
                              const float* __restrict__ agg,
                              unsigned short* __restrict__ xb) {
    size_t i = (size_t)blockIdx.x * blockDim.x + threadIdx.x;
    if (i >= (size_t)N_NODES * HID) return;
    float v = x[i] + agg[i];
    x[i]  = v;
    xb[i] = f2bf(v);
}

// ---------------------------------------------------------------------------
// out = x @ W_out + b_out   ([N,128] x [128,3], tiny)
// ---------------------------------------------------------------------------
__global__ void out_proj_kernel(const float* __restrict__ x,
                                const float* __restrict__ Wout,
                                const float* __restrict__ bout,
                                float* __restrict__ out) {
    int n = blockIdx.x * blockDim.x + threadIdx.x;
    if (n >= N_NODES) return;
    float a0 = bout[0], a1 = bout[1], a2 = bout[2];
    const float* xr = x + (size_t)n * HID;
#pragma unroll 8
    for (int k = 0; k < HID; ++k) {
        float xv = xr[k];
        a0 = fmaf(xv, Wout[k * 3 + 0], a0);
        a1 = fmaf(xv, Wout[k * 3 + 1], a1);
        a2 = fmaf(xv, Wout[k * 3 + 2], a2);
    }
    out[(size_t)n * 3 + 0] = a0;
    out[(size_t)n * 3 + 1] = a1;
    out[(size_t)n * 3 + 2] = a2;
}

// ---------------------------------------------------------------------------

extern "C" void kernel_launch(void* const* d_in, const int* in_sizes, int n_in,
                              void* d_out, int out_size, void* d_ws, size_t ws_size,
                              hipStream_t stream) {
    const float* h     = (const float*)d_in[0];
    const int*   eidx  = (const int*)  d_in[1];
    const float* eattr = (const float*)d_in[2];
    const float* Win   = (const float*)d_in[3];
    const float* bin   = (const float*)d_in[4];
    const float* Wmsg  = (const float*)d_in[5];
    const float* bmsg  = (const float*)d_in[6];
    const float* Wout  = (const float*)d_in[7];
    const float* bout  = (const float*)d_in[8];
    float* out = (float*)d_out;

    char* ws = (char*)d_ws;
    size_t off = 0;
    auto alloc = [&](size_t bytes) {
        char* p = ws + off;
        off = (off + bytes + 255) & ~(size_t)255;
        return p;
    };
    float*          x   = (float*)         alloc((size_t)N_NODES * HID * 4);
    float*          agg = (float*)         alloc((size_t)N_NODES * HID * 4);
    unsigned short* xb  = (unsigned short*)alloc((size_t)N_NODES * HID * 2);
    unsigned short* wp  = (unsigned short*)alloc((size_t)N_LAYERS * WP_U4 * 16);

    // Pre-swizzle message weights into WMMA B-fragment layout.
    {
        int total = N_LAYERS * KBLK * NTILES * 32;
        pack_wmsg_kernel<<<(total + 127) / 128, 128, 0, stream>>>(Wmsg, wp);
    }

    // Input projection (+ bf16 node features).
    input_proj_kernel<<<N_NODES, 128, 0, stream>>>(h, Win, bin, x, xb);

    for (int l = 0; l < N_LAYERS; ++l) {
        hipMemsetAsync(agg, 0, (size_t)N_NODES * HID * 4, stream);
        edge_msg_kernel<<<EBLOCKS, 256, 0, stream>>>(
            xb, eidx, eattr,
            wp + (size_t)l * WP_U4 * 8,       // uint4 -> 8 shorts
            bmsg + (size_t)l * HID, agg);
        int upd = (int)(((size_t)N_NODES * HID + 255) / 256);
        update_kernel<<<upd, 256, 0, stream>>>(x, agg, xb);
    }

    out_proj_kernel<<<(N_NODES + 255) / 256, 256, 0, stream>>>(x, Wout, bout, out);
}
